// Visual_update_5488968204376
// MI455X (gfx1250) — matrix operand order repaired and checked by reading, not compiled, not run on hardware
//
#include <hip/hip_runtime.h>
#include <hip/hip_bf16.h>

typedef __attribute__((ext_vector_type(16))) __bf16 v16bf;
typedef __attribute__((ext_vector_type(8)))  __bf16 v8bf;
typedef __attribute__((ext_vector_type(4)))  __bf16 v4bf;
typedef __attribute__((ext_vector_type(2)))  __bf16 v2bf;
typedef __attribute__((ext_vector_type(8)))  float  v8f;
typedef int v2i __attribute__((__vector_size__(8)));
typedef int v4i __attribute__((__vector_size__(16)));

// Problem constants
#define PP    36
#define NB    4
#define NRND  10
#define BNN   40          // NB*NRND
#define RELC  256
#define LSTMH 512
#define IMGF  2048

// Async global->LDS path (CDNA5). Guarded so absence cannot break the build.
#if defined(__has_builtin)
#  if __has_builtin(__builtin_amdgcn_global_load_async_to_lds_b128) && \
      __has_builtin(__builtin_amdgcn_global_load_async_to_lds_b64) &&  \
      __has_builtin(__builtin_amdgcn_s_wait_asynccnt)
#    define HAVE_ASYNC 1
#  endif
#endif
#ifndef HAVE_ASYNC
#  define HAVE_ASYNC 0
#endif
#define AS1 __attribute__((address_space(1)))
#define AS3 __attribute__((address_space(3)))

__device__ __forceinline__ void st4bf(__bf16* dst, float4 f) {
  v4bf p; p[0] = (__bf16)f.x; p[1] = (__bf16)f.y; p[2] = (__bf16)f.z; p[3] = (__bf16)f.w;
  *(v4bf*)dst = p;
}

// ---------------------------------------------------------------------------
// bf16 WMMA GEMM.  Block tile 32(M) x 128(N), K-step 32, 256 thr = 8 waves
// (2x4), wave tile 16x32 -> 2 WMMA / k-step.
//  A: bf16 row-major in global; fragment = 2 contiguous 16B LDS chunks.
//  B: bf16 pre-permuted in global to FRAGMENT-TILED layout: per (colblock,
//     kstep) an 8KB tile [nsub][lane][e]; staging is a raw contiguous copy.
//  Staging uses GLOBAL_LOAD_ASYNC_TO_LDS (ASYNCcnt) when available.
// ---------------------------------------------------------------------------
struct BufA { __bf16 raw[32 * 40]; };        // row stride 40 elems (80B)
struct BufB { __bf16 raw[8 * 768]; };        // nsub 1536B, lane stride 48B

__device__ __forceinline__ void stage_tiles(
    const __bf16* __restrict__ Ag, int lda, int rowBase, int k0,
    const __bf16* __restrict__ Btile, BufA& a, BufB& b)
{
  const int tid = threadIdx.x;
  // A: 32 rows x 64B; each thread one 8B granule
  {
    const int r = tid >> 3, q = tid & 7;
    const __bf16* src = Ag + (size_t)(rowBase + r) * lda + k0 + q * 4;
    __bf16* dst = &a.raw[r * 40 + q * 4];
#if HAVE_ASYNC
    __builtin_amdgcn_global_load_async_to_lds_b64((AS1 v2i*)src, (AS3 v2i*)dst, 0, 0);
#else
    *(v4bf*)dst = *(const v4bf*)src;
#endif
  }
  // B: 8KB contiguous tile; each thread two 16B granules (re-padded in LDS)
  #pragma unroll
  for (int h = 0; h < 2; ++h) {
    const int g  = tid * 2 + h;                 // 0..511
    const int ns = g >> 6, ln = (g >> 1) & 31, hf = g & 1;
    const __bf16* src = Btile + (size_t)g * 8;
    __bf16* dst = &b.raw[ns * 768 + ln * 24 + hf * 8];
#if HAVE_ASYNC
    __builtin_amdgcn_global_load_async_to_lds_b128((AS1 v4i*)src, (AS3 v4i*)dst, 0, 0);
#else
    *(v8bf*)dst = *(const v8bf*)src;
#endif
  }
}

__device__ __forceinline__ void gemm_core(
    const __bf16* __restrict__ Ag, int lda,
    const __bf16* __restrict__ Bt,   // fragment-tiled base for this col block
    int nsteps, int rowBase, BufA* sa, BufB* sb,
    v8f& acc0, v8f& acc1)
{
  const int tid  = threadIdx.x;
  const int lane = tid & 31;
  const int wave = tid >> 5;
  const int wm   = wave >> 2;     // 0..1
  const int wn4  = wave & 3;      // 0..3
  const int hb   = lane >> 4;

  v8f z = {};
  acc0 = z; acc1 = z;

  stage_tiles(Ag, lda, rowBase, 0, Bt, sa[0], sb[0]);
#if HAVE_ASYNC
  __builtin_amdgcn_s_wait_asynccnt(0);
#endif
  __syncthreads();

  for (int s = 0; s < nsteps; ++s) {
    BufA& ca = sa[s & 1];
    BufB& cb = sb[s & 1];
    if (s + 1 < nsteps)
      stage_tiles(Ag, lda, rowBase, (s + 1) << 5, Bt + (size_t)(s + 1) * 4096,
                  sa[(s + 1) & 1], sb[(s + 1) & 1]);
    if (s + 2 < nsteps)   // stream weight tiles toward L2 (global_prefetch_b8)
      __builtin_prefetch(Bt + (size_t)(s + 2) * 4096 + tid * 16, 0, 1);

    // fragments: contiguous 16B -> ds_load_b128
    const int am = (wm << 4) + (lane & 15);
    const v8bf alo = *(const v8bf*)&ca.raw[am * 40 + (hb << 4)];
    const v8bf ahi = *(const v8bf*)&ca.raw[am * 40 + (hb << 4) + 8];
    const v16bf af = __builtin_shufflevector(alo, ahi,
        0,1,2,3,4,5,6,7,8,9,10,11,12,13,14,15);

    const int ns0 = wn4 << 1;
    const v8bf b0l = *(const v8bf*)&cb.raw[ns0 * 768 + lane * 24];
    const v8bf b0h = *(const v8bf*)&cb.raw[ns0 * 768 + lane * 24 + 8];
    const v8bf b1l = *(const v8bf*)&cb.raw[(ns0 + 1) * 768 + lane * 24];
    const v8bf b1h = *(const v8bf*)&cb.raw[(ns0 + 1) * 768 + lane * 24 + 8];
    const v16bf bf0 = __builtin_shufflevector(b0l, b0h,
        0,1,2,3,4,5,6,7,8,9,10,11,12,13,14,15);
    const v16bf bf1 = __builtin_shufflevector(b1l, b1h,
        0,1,2,3,4,5,6,7,8,9,10,11,12,13,14,15);

    acc0 = __builtin_amdgcn_wmma_f32_16x16x32_bf16(false, af, false, bf0,
                                                   (short)0, acc0, false, false);
    acc1 = __builtin_amdgcn_wmma_f32_16x16x32_bf16(false, af, false, bf1,
                                                   (short)0, acc1, false, false);
#if HAVE_ASYNC
    __builtin_amdgcn_s_wait_asynccnt(0);
#endif
    __syncthreads();
  }
}

#define EPILOGUE_IDX()                                            \
  const int lane = threadIdx.x & 31, wave = threadIdx.x >> 5;     \
  const int mb   = rowBase + (((wave >> 2)) << 4) + ((lane >> 4) << 3); \
  const int col0 = colBase + ((wave & 3) << 5) + (lane & 15);     \
  const int col1 = col0 + 16;

// ---------------------------------------------------------------------------
// Weight permute+convert: f32 [K,N] row-major -> bf16 fragment-tiled
// tile(cb,s) base = (cb*nsteps+s)*4096 ; elem (k,n) -> nsub*512+lane*16+e
// ---------------------------------------------------------------------------
__global__ void k_cvt_fragtile(const float* __restrict__ W, int ldw,
                               __bf16* __restrict__ out)
{
  const int s = blockIdx.x, cb = blockIdx.y, nsteps = gridDim.x;
  const int tid = threadIdx.x;
  __bf16* tile = out + ((size_t)cb * nsteps + s) * 4096;
  const float* src = W + (size_t)(s * 32) * ldw + cb * 128;
  #pragma unroll
  for (int it = 0; it < 2; ++it) {
    const int t  = tid + (it << 8);
    const int kk = (t >> 5) << 1;
    const int nn = (t & 31) << 2;
    const float4 f0 = *(const float4*)(src + (size_t)kk * ldw + nn);
    const float4 f1 = *(const float4*)(src + (size_t)(kk + 1) * ldw + nn);
    const int e  = kk & 15;
    const int lb = ((kk >> 4) << 4) + (nn & 15);
    __bf16* base = tile + (nn >> 4) * 512;
    v2bf p;
    p[0] = (__bf16)f0.x; p[1] = (__bf16)f1.x; *(v2bf*)(base + (lb + 0) * 16 + e) = p;
    p[0] = (__bf16)f0.y; p[1] = (__bf16)f1.y; *(v2bf*)(base + (lb + 1) * 16 + e) = p;
    p[0] = (__bf16)f0.z; p[1] = (__bf16)f1.z; *(v2bf*)(base + (lb + 2) * 16 + e) = p;
    p[0] = (__bf16)f0.w; p[1] = (__bf16)f1.w; *(v2bf*)(base + (lb + 3) * 16 + e) = p;
  }
}

// f32 -> bf16 row-major (8 elems / thread)
__global__ void k_cvt_bf(const float* __restrict__ in, __bf16* __restrict__ out)
{
  const size_t idx = ((size_t)blockIdx.x * 256 + threadIdx.x) * 8;
  st4bf(out + idx,     *(const float4*)(in + idx));
  st4bf(out + idx + 4, *(const float4*)(in + idx + 4));
}

// ---------------------------------------------------------------------------
// K1: rel = sigmoid(relation @ Wcr + bcr) -> bf16 [NB*PP*PP, RELC]
// ---------------------------------------------------------------------------
__global__ void k_rel(const float* __restrict__ relation, const float* __restrict__ Wcr,
                      const float* __restrict__ bcr, __bf16* __restrict__ relb)
{
  const int pos = blockIdx.x;
  const int c   = threadIdx.x;
  const float* rp = relation + (size_t)pos * 7;
  float s = bcr[c];
  #pragma unroll
  for (int d = 0; d < 7; ++d) s += rp[d] * Wcr[d * RELC + c];
  relb[(size_t)pos * RELC + c] = (__bf16)(1.0f / (1.0f + __expf(-s)));
}

// ---------------------------------------------------------------------------
// K2: sig = sigmoid(ques @ Wq + bq); v = sig*Wrw, u = sig*Wg
// ---------------------------------------------------------------------------
__global__ void k_ques(const float* __restrict__ ques, const float* __restrict__ Wq,
                       const float* __restrict__ bq, const float* __restrict__ Wrw,
                       const float* __restrict__ Wg, float* __restrict__ v,
                       float* __restrict__ u)
{
  const int bn = blockIdx.x;
  const int c  = threadIdx.x;
  const float* qp = ques + (size_t)bn * LSTMH;
  float s = bq[c];
  for (int k = 0; k < LSTMH; ++k) s += qp[k] * Wq[k * RELC + c];
  const float sig = 1.0f / (1.0f + __expf(-s));
  v[bn * RELC + c] = sig * Wrw[c];
  u[bn * RELC + c] = sig * Wg[c];
}

// ---------------------------------------------------------------------------
// K3: relWc = rel @ Wc[:256,:]   (M=5184, K=256, N=256)
// ---------------------------------------------------------------------------
__global__ void gemm_relWc(const __bf16* __restrict__ relb, const __bf16* __restrict__ Wt,
                           float* __restrict__ relWc)
{
  __shared__ BufA sa[2]; __shared__ BufB sb[2];
  const int rowBase = blockIdx.x * 32, colBase = blockIdx.y * 128;
  const int nsteps = RELC / 32;
  v8f a0, a1;
  gemm_core(relb, RELC, Wt + (size_t)blockIdx.y * nsteps * 4096, nsteps, rowBase, sa, sb, a0, a1);
  EPILOGUE_IDX();
  #pragma unroll
  for (int r = 0; r < 8; ++r) {
    relWc[(size_t)(mb + r) * RELC + col0] = a0[r];
    relWc[(size_t)(mb + r) * RELC + col1] = a1[r];
  }
}

// ---------------------------------------------------------------------------
// K4: imgWc = img @ Wc[256:,:] + bc   (M=1440, K=2048, N=256)
// ---------------------------------------------------------------------------
__global__ void gemm_imgWc(const __bf16* __restrict__ imgb, const __bf16* __restrict__ Wt,
                           const float* __restrict__ bc, float* __restrict__ imgWc)
{
  __shared__ BufA sa[2]; __shared__ BufB sb[2];
  const int rowBase = blockIdx.x * 32, colBase = blockIdx.y * 128;
  const int nsteps = IMGF / 32;
  v8f a0, a1;
  gemm_core(imgb, IMGF, Wt + (size_t)blockIdx.y * nsteps * 4096, nsteps, rowBase, sa, sb, a0, a1);
  EPILOGUE_IDX();
  const float b0 = bc[col0], b1 = bc[col1];
  #pragma unroll
  for (int r = 0; r < 8; ++r) {
    imgWc[(size_t)(mb + r) * RELC + col0] = a0[r] + b0;
    imgWc[(size_t)(mb + r) * RELC + col1] = a1[r] + b1;
  }
}

// ---------------------------------------------------------------------------
// K5: attention. One block per (bn,i): rw softmax over j, then gw softmax.
// ---------------------------------------------------------------------------
__global__ void k_attn(const __bf16* __restrict__ relb, const float* __restrict__ relWc,
                       const float* __restrict__ imgWc, const float* __restrict__ v,
                       const float* __restrict__ u, const float* __restrict__ brw,
                       const float* __restrict__ bg, float* __restrict__ gw)
{
  const int bi = blockIdx.x;            // bn*36 + i
  const int bn = bi / PP, i = bi - bn * PP;
  const int b  = bn / NRND, q = bn - b * NRND;
  const int isrc = (q * PP + i) / NRND; // torch tile+view row scramble
  const int tid = threadIdx.x, lane = tid & 31, wave = tid >> 5;

  __shared__ float red[8];
  __shared__ float lw[PP];
  __shared__ float lg[PP];

  const float* vv = v + (size_t)bn * RELC;
  const float* uu = u + (size_t)bn * RELC;
  const size_t relRow = ((size_t)(b * PP + isrc) * PP) * RELC;
  const size_t imgRow = ((size_t)bn * PP) * RELC;

  for (int j = 0; j < PP; ++j) {
    float p = (float)relb[relRow + (size_t)j * RELC + tid] * vv[tid];
    #pragma unroll
    for (int o = 16; o > 0; o >>= 1) p += __shfl_down(p, o, 32);
    if (lane == 0) red[wave] = p;
    __syncthreads();
    if (tid == 0) {
      float s = 0.f;
      #pragma unroll
      for (int wv = 0; wv < 8; ++wv) s += red[wv];
      lw[j] = s + brw[0];
    }
    __syncthreads();
  }
  if (tid == 0) {
    float m = lw[0];
    for (int j = 1; j < PP; ++j) m = fmaxf(m, lw[j]);
    float s = 0.f;
    for (int j = 0; j < PP; ++j) { lw[j] = __expf(lw[j] - m); s += lw[j]; }
    const float inv = 1.0f / s;
    for (int j = 0; j < PP; ++j) lw[j] *= inv;
  }
  __syncthreads();

  for (int j = 0; j < PP; ++j) {
    const float p1 = relWc[relRow + (size_t)j * RELC + tid] * uu[tid];
    const float p2 = imgWc[imgRow + (size_t)j * RELC + tid] * uu[tid];
    float p = lw[j] * p1 + p2;
    #pragma unroll
    for (int o = 16; o > 0; o >>= 1) p += __shfl_down(p, o, 32);
    if (lane == 0) red[wave] = p;
    __syncthreads();
    if (tid == 0) {
      float s = 0.f;
      #pragma unroll
      for (int wv = 0; wv < 8; ++wv) s += red[wv];
      lg[j] = s + bg[0];
    }
    __syncthreads();
  }
  if (tid == 0) {
    float m = lg[0];
    for (int j = 1; j < PP; ++j) m = fmaxf(m, lg[j]);
    float s = 0.f;
    for (int j = 0; j < PP; ++j) { lg[j] = __expf(lg[j] - m); s += lg[j]; }
    const float inv = 1.0f / s;
    for (int j = 0; j < PP; ++j) lg[j] *= inv;
  }
  __syncthreads();
  if (tid < PP) gw[(size_t)bi * PP + tid] = lg[tid];
}

// ---------------------------------------------------------------------------
// K6: cii_bf[row,:2048] = img[row]; cii_bf[row,2048:] = sum_j gw*img[bn,j,:]
// ---------------------------------------------------------------------------
__global__ void k_ws(const float* __restrict__ img, const float* __restrict__ gw,
                     __bf16* __restrict__ ciib)
{
  const int bi = blockIdx.x;
  const int bn = bi / PP;
  const int tid = threadIdx.x;
  __shared__ float w[PP];
  if (tid < PP) w[tid] = gw[(size_t)bi * PP + tid];
  __syncthreads();

  const float* imgbn = img + (size_t)bn * PP * IMGF;
  float acc[8];
  #pragma unroll
  for (int t = 0; t < 8; ++t) acc[t] = 0.f;
  for (int j = 0; j < PP; ++j) {
    const float wj = w[j];
    const float* row = imgbn + (size_t)j * IMGF;
    #pragma unroll
    for (int t = 0; t < 8; ++t) acc[t] += wj * row[tid + t * 256];
  }
  __bf16* crow = ciib + (size_t)bi * (2 * IMGF);
  const float* myimg = img + (size_t)bi * IMGF;
  #pragma unroll
  for (int t = 0; t < 8; ++t) {
    crow[tid + t * 256]        = (__bf16)myimg[tid + t * 256];
    crow[IMGF + tid + t * 256] = (__bf16)acc[t];
  }
}

// ---------------------------------------------------------------------------
// K7: gated_bf = (bf16)( sigmoid(cii @ Ww + bw) * cii )   (M=1440, K=N=4096)
// ---------------------------------------------------------------------------
__global__ void gemm_gate(const __bf16* __restrict__ ciib, const __bf16* __restrict__ Wt,
                          const float* __restrict__ bw, __bf16* __restrict__ gatedb)
{
  __shared__ BufA sa[2]; __shared__ BufB sb[2];
  const int rowBase = blockIdx.x * 32, colBase = blockIdx.y * 128;
  const int nsteps = 2 * IMGF / 32;
  v8f a0, a1;
  gemm_core(ciib, 2 * IMGF, Wt + (size_t)blockIdx.y * nsteps * 4096, nsteps, rowBase, sa, sb, a0, a1);
  EPILOGUE_IDX();
  const float b0 = bw[col0], b1 = bw[col1];
  #pragma unroll
  for (int r = 0; r < 8; ++r) {
    const size_t i0 = (size_t)(mb + r) * (2 * IMGF) + col0;
    const size_t i1 = (size_t)(mb + r) * (2 * IMGF) + col1;
    gatedb[i0] = (__bf16)((float)ciib[i0] / (1.0f + __expf(-(a0[r] + b0))));
    gatedb[i1] = (__bf16)((float)ciib[i1] / (1.0f + __expf(-(a1[r] + b1))));
  }
}

// ---------------------------------------------------------------------------
// K8: out = gated @ Wi + bi   (M=1440, K=4096, N=2048)
// ---------------------------------------------------------------------------
__global__ void gemm_out(const __bf16* __restrict__ gatedb, const __bf16* __restrict__ Wt,
                         const float* __restrict__ bi, float* __restrict__ out)
{
  __shared__ BufA sa[2]; __shared__ BufB sb[2];
  const int rowBase = blockIdx.x * 32, colBase = blockIdx.y * 128;
  const int nsteps = 2 * IMGF / 32;
  v8f a0, a1;
  gemm_core(gatedb, 2 * IMGF, Wt + (size_t)blockIdx.y * nsteps * 4096, nsteps, rowBase, sa, sb, a0, a1);
  EPILOGUE_IDX();
  const float b0 = bi[col0], b1 = bi[col1];
  #pragma unroll
  for (int r = 0; r < 8; ++r) {
    out[(size_t)(mb + r) * IMGF + col0] = a0[r] + b0;
    out[(size_t)(mb + r) * IMGF + col1] = a1[r] + b1;
  }
}

// ---------------------------------------------------------------------------
extern "C" void kernel_launch(void* const* d_in, const int* in_sizes, int n_in,
                              void* d_out, int out_size, void* d_ws, size_t ws_size,
                              hipStream_t stream)
{
  (void)in_sizes; (void)n_in; (void)out_size; (void)ws_size;
  const float* relation = (const float*)d_in[0];
  const float* img  = (const float*)d_in[1];
  const float* ques = (const float*)d_in[2];
  const float* Wcr  = (const float*)d_in[3];
  const float* bcr  = (const float*)d_in[4];
  const float* Wq   = (const float*)d_in[5];
  const float* bq   = (const float*)d_in[6];
  const float* Wrw  = (const float*)d_in[7];
  const float* brw  = (const float*)d_in[8];
  const float* Wc   = (const float*)d_in[9];
  const float* bc   = (const float*)d_in[10];
  const float* Wg   = (const float*)d_in[11];
  const float* bg   = (const float*)d_in[12];
  const float* Ww   = (const float*)d_in[13];
  const float* bw   = (const float*)d_in[14];
  const float* Wi   = (const float*)d_in[15];
  const float* bi   = (const float*)d_in[16];
  float* out = (float*)d_out;

  char* wsp = (char*)d_ws;
  size_t off = 0;
  auto alloc = [&](size_t bytes) -> void* {
    void* p = (void*)(wsp + off);
    off += (bytes + 255) & ~(size_t)255;
    return p;
  };
  __bf16* relb   = (__bf16*)alloc((size_t)NB * PP * PP * RELC * 2);   // 2.65 MB
  float*  vq     = (float*) alloc((size_t)BNN * RELC * 4);
  float*  ug     = (float*) alloc((size_t)BNN * RELC * 4);
  float*  relWc  = (float*) alloc((size_t)NB * PP * PP * RELC * 4);   // 5.3 MB
  float*  imgWc  = (float*) alloc((size_t)BNN * PP * RELC * 4);       // 1.5 MB
  float*  gw     = (float*) alloc((size_t)BNN * PP * PP * 4);         // 0.2 MB
  __bf16* imgb   = (__bf16*)alloc((size_t)BNN * PP * IMGF * 2);       // 5.9 MB
  __bf16* ciib   = (__bf16*)alloc((size_t)BNN * PP * 2 * IMGF * 2);   // 11.8 MB
  __bf16* gatedb = (__bf16*)alloc((size_t)BNN * PP * 2 * IMGF * 2);   // 11.8 MB
  __bf16* Wc1t   = (__bf16*)alloc((size_t)RELC * RELC * 2);           // 0.13 MB
  __bf16* Wc2t   = (__bf16*)alloc((size_t)IMGF * RELC * 2);           // 1.0 MB
  __bf16* Wwt    = (__bf16*)alloc((size_t)(2*IMGF) * (2*IMGF) * 2);   // 33.6 MB
  __bf16* Wit    = (__bf16*)alloc((size_t)(2*IMGF) * IMGF * 2);       // 16.8 MB

  // ---- one-time (per launch) conversions / permutes ----
  k_cvt_fragtile<<<dim3(RELC / 32,       RELC / 128),       dim3(256), 0, stream>>>(Wc, RELC, Wc1t);
  k_cvt_fragtile<<<dim3(IMGF / 32,       RELC / 128),       dim3(256), 0, stream>>>(Wc + (size_t)RELC * RELC, RELC, Wc2t);
  k_cvt_fragtile<<<dim3(2 * IMGF / 32,   2 * IMGF / 128),   dim3(256), 0, stream>>>(Ww, 2 * IMGF, Wwt);
  k_cvt_fragtile<<<dim3(2 * IMGF / 32,   IMGF / 128),       dim3(256), 0, stream>>>(Wi, IMGF, Wit);
  k_cvt_bf<<<dim3((BNN * PP * IMGF) / 2048), dim3(256), 0, stream>>>(img, imgb);

  // ---- pipeline ----
  k_rel  <<<dim3(NB * PP * PP), dim3(256), 0, stream>>>(relation, Wcr, bcr, relb);
  k_ques <<<dim3(BNN),          dim3(256), 0, stream>>>(ques, Wq, bq, Wrw, Wg, vq, ug);
  gemm_relWc<<<dim3((NB * PP * PP) / 32, RELC / 128), dim3(256), 0, stream>>>(relb, Wc1t, relWc);
  gemm_imgWc<<<dim3((BNN * PP) / 32, RELC / 128), dim3(256), 0, stream>>>(imgb, Wc2t, bc, imgWc);
  k_attn <<<dim3(BNN * PP), dim3(256), 0, stream>>>(relb, relWc, imgWc, vq, ug, brw, bg, gw);
  k_ws   <<<dim3(BNN * PP), dim3(256), 0, stream>>>(img, gw, ciib);
  gemm_gate<<<dim3((BNN * PP) / 32, (2 * IMGF) / 128), dim3(256), 0, stream>>>(ciib, Wwt, bw, gatedb);
  gemm_out <<<dim3((BNN * PP) / 32, IMGF / 128), dim3(256), 0, stream>>>(gatedb, Wit, bi, out);
}
